// GCNGraph_28372553957768
// MI455X (gfx1250) — matrix operand order repaired and checked by reading, not compiled, not run on hardware
//
#include <hip/hip_runtime.h>
#include <hip/hip_bf16.h>

// ---------------------------------------------------------------------------
// CDNA5 / gfx1250 GCN pipeline: bf16 WMMA GEMMs, fp32 accumulate,
// async global->LDS staging (double-buffered) for the HBM-bound adjacency GEMM.
// ---------------------------------------------------------------------------

typedef __attribute__((ext_vector_type(16))) __bf16 v16bf;
typedef __attribute__((ext_vector_type(8)))  float  v8f;
typedef __attribute__((ext_vector_type(4)))  int    v4i;

#define NNODES   8192
#define INFEATS  256
#define HFEATS   512
#define NGRAPHS  64

#if defined(__has_builtin)
#  if __has_builtin(__builtin_amdgcn_global_load_async_to_lds_b128) && \
      __has_builtin(__builtin_amdgcn_s_wait_asynccnt)
#    define HAVE_ASYNC 1
#  endif
#endif
#ifndef HAVE_ASYNC
#  define HAVE_ASYNC 0
#endif

#if HAVE_ASYNC
typedef __attribute__((address_space(1))) v4i gv4i;   // global 16B chunk
typedef __attribute__((address_space(3))) v4i lv4i;   // LDS 16B chunk
#endif

__device__ __forceinline__ unsigned short f2bf(float f) {
    unsigned int u = __float_as_uint(f);
    u = u + 0x7FFFu + ((u >> 16) & 1u);   // round-to-nearest-even
    return (unsigned short)(u >> 16);
}
__device__ __forceinline__ unsigned int pack2(float a, float b) {
    return (unsigned int)f2bf(a) | ((unsigned int)f2bf(b) << 16);
}

union FragBF { v16bf bf; uint4 u[2]; };

// 16-byte global -> LDS copy; async DMA (ASYNCcnt) when available.
__device__ __forceinline__ void cp16_async(void* l, const void* g) {
#if HAVE_ASYNC
    gv4i* gp = (gv4i*)(unsigned long long)(g);
    lv4i* lp = (lv4i*)(unsigned int)(unsigned long long)(l);
    __builtin_amdgcn_global_load_async_to_lds_b128(gp, lp, 0, 0);
#else
    *reinterpret_cast<uint4*>(l) = *reinterpret_cast<const uint4*>(g);
#endif
}
__device__ __forceinline__ void wait_async_keep2() {
#if HAVE_ASYNC
    __builtin_amdgcn_s_wait_asynccnt(2);
#endif
}
__device__ __forceinline__ void wait_async_all() {
#if HAVE_ASYNC
    __builtin_amdgcn_s_wait_asynccnt(0);
#endif
}

// ---------------------------------------------------------------------------
// Transpose+convert W [K,512] f32 -> Wt [512][K] bf16
// ---------------------------------------------------------------------------
__global__ void wt_kernel(const float* __restrict__ W, unsigned short* __restrict__ Wt, int Kin) {
    int idx = blockIdx.x * blockDim.x + threadIdx.x;
    int total = Kin * HFEATS;
    if (idx < total) {
        int n = idx / Kin;
        int k = idx - n * Kin;
        Wt[idx] = f2bf(W[k * HFEATS + n]);
    }
}

// ---------------------------------------------------------------------------
// Generic WMMA GEMM for the X@W stages:  C[m,n] = sum_k A[m,k] * Bt[n,k]
//   A: row-major [M,K], f32 (AF32) or raw bf16; Bt: row-major [N,K] bf16
//   Out: bf16 [N][ldout] transposed (feeds adjacency GEMM as B operand)
// ---------------------------------------------------------------------------
template<int MB, int NB, bool AF32>
__global__ __launch_bounds__(256)
void gemm_bf16_wmma(const void* __restrict__ Aq,
                    const unsigned short* __restrict__ Bt,
                    void* __restrict__ Out,
                    int M, int N, int K, int ldout) {
    constexpr int MT      = MB / 16;
    constexpr int WAVES_N = 8 / MT;
    constexpr int NT      = NB / 16;
    constexpr int NT_W    = NT / WAVES_N;
    constexpr int LDSS    = 80;

    __shared__ __align__(16) unsigned char smem[(MB + NB) * LDSS];
    unsigned char* As = smem;
    unsigned char* Bs = smem + MB * LDSS;

    const int tid  = threadIdx.x;
    const int wave = tid >> 5;
    const int lane = tid & 31;
    const int m0   = blockIdx.x * MB;
    const int n0   = blockIdx.y * NB;

    const int mt   = wave % MT;
    const int nt0  = (wave / MT) * NT_W;
    const int lrow = lane & 15;
    const int lhal = lane >> 4;

    const int lr = tid >> 2;          // 0..63
    const int lc = (tid & 3) * 8;     // 0,8,16,24

    v8f acc[NT_W] = {};

    const float*          Af = (const float*)Aq;
    const unsigned short* Ab = (const unsigned short*)Aq;

    for (int k0 = 0; k0 < K; k0 += 32) {
        if (k0) __syncthreads();

        #pragma unroll
        for (int p = 0; p < MB / 64; ++p) {
            int row = lr + p * 64;
            size_t go = (size_t)(m0 + row) * K + k0 + lc;
            uint4 o;
            if (AF32) {
                const float4* pa = reinterpret_cast<const float4*>(Af + go);
                float4 x = pa[0], y = pa[1];
                o.x = pack2(x.x, x.y); o.y = pack2(x.z, x.w);
                o.z = pack2(y.x, y.y); o.w = pack2(y.z, y.w);
                if (k0 + 32 < K) __builtin_prefetch(Af + go + 32, 0, 1);
            } else {
                o = *reinterpret_cast<const uint4*>(Ab + go);
                if (k0 + 32 < K) __builtin_prefetch(Ab + go + 32, 0, 1);
            }
            *reinterpret_cast<uint4*>(As + row * LDSS + lc * 2) = o;
        }

        #pragma unroll
        for (int p = 0; p < NB / 64; ++p) {
            int row = lr + p * 64;
            size_t go = (size_t)(n0 + row) * K + k0 + lc;
            uint4 v = *reinterpret_cast<const uint4*>(Bt + go);
            if (k0 + 32 < K) __builtin_prefetch(Bt + go + 32, 0, 1);
            *reinterpret_cast<uint4*>(Bs + row * LDSS + lc * 2) = v;
        }

        __syncthreads();

        FragBF fa;
        {
            const unsigned char* base = As + (mt * 16 + lrow) * LDSS;
            fa.u[0] = *reinterpret_cast<const uint4*>(base + lhal * 16);
            fa.u[1] = *reinterpret_cast<const uint4*>(base + lhal * 16 + 32);
        }

        #pragma unroll
        for (int j = 0; j < NT_W; ++j) {
            FragBF fb;
            const unsigned char* base = Bs + ((nt0 + j) * 16 + lrow) * LDSS + lhal * 16;
            fb.u[0] = *reinterpret_cast<const uint4*>(base);
            fb.u[1] = *reinterpret_cast<const uint4*>(base + 32);
            acc[j] = __builtin_amdgcn_wmma_f32_16x16x32_bf16(
                        false, fa.bf, false, fb.bf, (short)0, acc[j], false, false);
        }
    }

    // transposed bf16 store: Out[n][m]
    #pragma unroll
    for (int j = 0; j < NT_W; ++j) {
        int n = n0 + (nt0 + j) * 16 + lrow;
        #pragma unroll
        for (int r = 0; r < 8; ++r) {
            int m = m0 + mt * 16 + r + 8 * lhal;
            ((unsigned short*)Out)[(size_t)n * ldout + m] = f2bf(acc[j][r]);
        }
    }
}

// ---------------------------------------------------------------------------
// Adjacency GEMM:  H[m,n] = relu?( sum_k adj[m,k] * XWt[n,k] + bias[n] )
//   adj f32 [8192,8192] streamed from HBM -> async DMA into double-buffered
//   raw-f32 LDS tiles (converted to bf16 at fragment build).
//   XWt bf16 [512][8192] (L2-resident) -> async DMA, single buffer.
//   MB=64 rows per WG, full N=512 -> adj is read exactly once per layer.
// ---------------------------------------------------------------------------
template<bool OUTF32, bool RELU>
__global__ __launch_bounds__(256)
void adj_gemm_wmma(const float* __restrict__ adj,
                   const unsigned short* __restrict__ Bt,
                   const float* __restrict__ bias,
                   void* __restrict__ Out) {
    constexpr int MB   = 64;
    constexpr int NT_W = 16;             // 8 waves: 4 in m, 2 in n (16 tiles each)
    constexpr int ASTR = 144;            // A row: 32 f32 (128B) + 16B pad
    constexpr int BSTR = 80;             // B row: 32 bf16 (64B) + 16B pad
    constexpr int ABYTES = MB * ASTR;    // 9216

    __shared__ __align__(16) unsigned char smem[2 * ABYTES + HFEATS * BSTR]; // 59392 B
    unsigned char* Bs = smem + 2 * ABYTES;

    const int tid  = threadIdx.x;
    const int wave = tid >> 5;
    const int lane = tid & 31;
    const int m0   = blockIdx.x * MB;

    const int mt   = wave & 3;
    const int nt0  = (wave >> 2) * NT_W;
    const int lrow = lane & 15;
    const int lhal = lane >> 4;

    // A loader: row = tid/4 (0..63), 32B segment = (tid&3)
    const int ar = tid >> 2;
    const int ac = (tid & 3) * 8;        // f32 element offset (8 f32 = 32B)

    v8f acc[NT_W] = {};

    // prologue: stage A tile(k=0) into buffer 0
    {
        const float* g = adj + (size_t)(m0 + ar) * NNODES + ac;
        unsigned char* l = smem + ar * ASTR + ac * 4;
        cp16_async(l,      g);
        cp16_async(l + 16, (const char*)g + 16);
    }

    for (int k0 = 0, it = 0; k0 < NNODES; k0 += 32, ++it) {
        unsigned char* Acur = smem + (it & 1) * ABYTES;
        unsigned char* Anxt = smem + ((it & 1) ^ 1) * ABYTES;

        __syncthreads();   // previous compute done: safe to refill Bs / Anxt

        // issue B(k0): 2 rows per thread, 4 x 16B per row
        {
            int r0 = tid * 2;
            #pragma unroll
            for (int rr = 0; rr < 2; ++rr) {
                const unsigned short* g = Bt + (size_t)(r0 + rr) * NNODES + k0;
                unsigned char* l = Bs + (r0 + rr) * BSTR;
                #pragma unroll
                for (int s = 0; s < 4; ++s)
                    cp16_async(l + s * 16, (const char*)g + s * 16);
            }
        }
        // issue A(k0+32) into the other buffer (issued LAST so we can wait past B)
        const bool more = (k0 + 32) < NNODES;
        if (more) {
            const float* g = adj + (size_t)(m0 + ar) * NNODES + (k0 + 32) + ac;
            unsigned char* l = Anxt + ar * ASTR + ac * 4;
            cp16_async(l,      g);
            cp16_async(l + 16, (const char*)g + 16);
        }
        if (more) wait_async_keep2();   // in-order: B(k0)+A(k0) done, next-A in flight
        else      wait_async_all();
        __syncthreads();                // all lanes' tiles resident

        // A fragment: convert raw f32 tile -> bf16 (lane<16: K{0-7,16-23}, hi: +8)
        FragBF fa;
        {
            const unsigned char* base = Acur + (mt * 16 + lrow) * ASTR + lhal * 32;
            float4 x0 = *reinterpret_cast<const float4*>(base);
            float4 x1 = *reinterpret_cast<const float4*>(base + 16);
            float4 y0 = *reinterpret_cast<const float4*>(base + 64);
            float4 y1 = *reinterpret_cast<const float4*>(base + 80);
            uint4 u0, u1;
            u0.x = pack2(x0.x, x0.y); u0.y = pack2(x0.z, x0.w);
            u0.z = pack2(x1.x, x1.y); u0.w = pack2(x1.z, x1.w);
            u1.x = pack2(y0.x, y0.y); u1.y = pack2(y0.z, y0.w);
            u1.z = pack2(y1.x, y1.y); u1.w = pack2(y1.z, y1.w);
            fa.u[0] = u0; fa.u[1] = u1;
        }

        #pragma unroll
        for (int j = 0; j < NT_W; ++j) {
            FragBF fb;
            const unsigned char* base = Bs + ((nt0 + j) * 16 + lrow) * BSTR + lhal * 16;
            fb.u[0] = *reinterpret_cast<const uint4*>(base);
            fb.u[1] = *reinterpret_cast<const uint4*>(base + 32);
            acc[j] = __builtin_amdgcn_wmma_f32_16x16x32_bf16(
                        false, fa.bf, false, fb.bf, (short)0, acc[j], false, false);
        }
    }

    // epilogue: bias + optional relu, row-major out [M][512]
    #pragma unroll
    for (int j = 0; j < NT_W; ++j) {
        int n = (nt0 + j) * 16 + lrow;
        float bv = bias[n];
        #pragma unroll
        for (int r = 0; r < 8; ++r) {
            int m = m0 + mt * 16 + r + 8 * lhal;
            float v = acc[j][r] + bv;
            if (RELU) v = v > 0.0f ? v : 0.0f;
            if (OUTF32) ((float*)Out)[(size_t)m * HFEATS + n] = v;
            else        ((unsigned short*)Out)[(size_t)m * HFEATS + n] = f2bf(v);
        }
    }
}

// ---------------------------------------------------------------------------
// Per-graph mean readout: ids sorted -> binary-search segment bounds
// ---------------------------------------------------------------------------
__global__ void seg_mean_kernel(const float* __restrict__ H, const int* __restrict__ ids,
                                float* __restrict__ g) {
    int gph = blockIdx.x;
    int lo = 0, hi = NNODES;
    while (lo < hi) { int mid = (lo + hi) >> 1; if (ids[mid] < gph) lo = mid + 1; else hi = mid; }
    int start = lo;
    lo = start; hi = NNODES;
    while (lo < hi) { int mid = (lo + hi) >> 1; if (ids[mid] < gph + 1) lo = mid + 1; else hi = mid; }
    int end = lo;
    float inv = 1.0f / (float)((end - start) > 0 ? (end - start) : 1);
    for (int c = threadIdx.x; c < HFEATS; c += blockDim.x) {
        float s = 0.0f;
        for (int i = start; i < end; ++i) s += H[(size_t)i * HFEATS + c];
        g[gph * HFEATS + c] = s * inv;
    }
}

// ---------------------------------------------------------------------------
// Tiny MLP tail: [64,512] -> 16 relu -> 8 relu -> 1 sigmoid
// ---------------------------------------------------------------------------
__global__ void mlp_kernel(const float* __restrict__ g,
                           const float* __restrict__ D1, const float* __restrict__ db1,
                           const float* __restrict__ D2, const float* __restrict__ db2,
                           const float* __restrict__ D3, const float* __restrict__ db3,
                           float* __restrict__ out) {
    int i = threadIdx.x;
    if (i >= NGRAPHS) return;
    float h1[16];
    for (int n = 0; n < 16; ++n) {
        float s = db1[n];
        for (int k = 0; k < HFEATS; ++k) s += g[i * HFEATS + k] * D1[k * 16 + n];
        h1[n] = s > 0.0f ? s : 0.0f;
    }
    float h2[8];
    for (int n = 0; n < 8; ++n) {
        float s = db2[n];
        for (int k = 0; k < 16; ++k) s += h1[k] * D2[k * 8 + n];
        h2[n] = s > 0.0f ? s : 0.0f;
    }
    float s = db3[0];
    for (int k = 0; k < 8; ++k) s += h2[k] * D3[k];
    out[i] = 1.0f / (1.0f + expf(-s));
}

// ---------------------------------------------------------------------------
extern "C" void kernel_launch(void* const* d_in, const int* in_sizes, int n_in,
                              void* d_out, int out_size, void* d_ws, size_t ws_size,
                              hipStream_t stream) {
    (void)in_sizes; (void)n_in; (void)out_size; (void)ws_size;
    const float* in_feat  = (const float*)d_in[0];
    const float* e_weight = (const float*)d_in[1];
    const int*   seg_ids  = (const int*)d_in[2];
    const float* W1 = (const float*)d_in[4];  const float* b1 = (const float*)d_in[5];
    const float* W2 = (const float*)d_in[6];  const float* b2 = (const float*)d_in[7];
    const float* W3 = (const float*)d_in[8];  const float* b3 = (const float*)d_in[9];
    const float* D1 = (const float*)d_in[10]; const float* db1 = (const float*)d_in[11];
    const float* D2 = (const float*)d_in[12]; const float* db2 = (const float*)d_in[13];
    const float* D3 = (const float*)d_in[14]; const float* db3 = (const float*)d_in[15];

    char* ws = (char*)d_ws;
    unsigned short* Wt1 = (unsigned short*)ws;  ws += (size_t)HFEATS * INFEATS * 2;
    unsigned short* Wt2 = (unsigned short*)ws;  ws += (size_t)HFEATS * HFEATS * 2;
    unsigned short* Wt3 = (unsigned short*)ws;  ws += (size_t)HFEATS * HFEATS * 2;
    unsigned short* XWt = (unsigned short*)ws;  ws += (size_t)HFEATS * NNODES * 2;
    unsigned short* Hbf = (unsigned short*)ws;  ws += (size_t)NNODES * HFEATS * 2;
    float*          H3  = (float*)ws;           ws += (size_t)NNODES * HFEATS * 4;
    float*          gbuf= (float*)ws;           ws += (size_t)NGRAPHS * HFEATS * 4;

    wt_kernel<<<(HFEATS * INFEATS + 255) / 256, 256, 0, stream>>>(W1, Wt1, INFEATS);
    wt_kernel<<<(HFEATS * HFEATS + 255) / 256, 256, 0, stream>>>(W2, Wt2, HFEATS);
    wt_kernel<<<(HFEATS * HFEATS + 255) / 256, 256, 0, stream>>>(W3, Wt3, HFEATS);

    dim3 blk(256);
    dim3 gXW(NNODES / 128, HFEATS / 128);   // 64 x 4
    dim3 gADJ(NNODES / 64, 1);              // 128 x 1

    // layer 1
    gemm_bf16_wmma<128, 128, true><<<gXW, blk, 0, stream>>>(
        in_feat, Wt1, XWt, NNODES, HFEATS, INFEATS, NNODES);
    adj_gemm_wmma<false, true><<<gADJ, blk, 0, stream>>>(e_weight, XWt, b1, Hbf);
    // layer 2
    gemm_bf16_wmma<128, 128, false><<<gXW, blk, 0, stream>>>(
        Hbf, Wt2, XWt, NNODES, HFEATS, HFEATS, NNODES);
    adj_gemm_wmma<false, true><<<gADJ, blk, 0, stream>>>(e_weight, XWt, b2, Hbf);
    // layer 3 (no relu, f32 out for readout)
    gemm_bf16_wmma<128, 128, false><<<gXW, blk, 0, stream>>>(
        Hbf, Wt3, XWt, NNODES, HFEATS, HFEATS, NNODES);
    adj_gemm_wmma<true, false><<<gADJ, blk, 0, stream>>>(e_weight, XWt, b3, H3);

    // readout + MLP tail
    seg_mean_kernel<<<NGRAPHS, 256, 0, stream>>>(H3, seg_ids, gbuf);
    mlp_kernel<<<1, 64, 0, stream>>>(gbuf, D1, db1, D2, db2, D3, db3, (float*)d_out);
}
